// LlamaDecoderLayer_65798898975426
// MI455X (gfx1250) — matrix operand order repaired and checked by reading, not compiled, not run on hardware
//
#include <hip/hip_runtime.h>
#include <stdint.h>

// ---------------------------------------------------------------------------
// CDNA5 (gfx1250) Llama decoder layer, bf16 WMMA (16x16x32, f32 accum),
// TDM (tensor_load_to_lds) double-buffered A-tile staging in the GEMMs.
// ---------------------------------------------------------------------------

typedef __attribute__((ext_vector_type(16))) __bf16 v16bf;
typedef __attribute__((ext_vector_type(8)))  float  v8f;
typedef __attribute__((ext_vector_type(4)))  unsigned int u32x4_t;
typedef __attribute__((ext_vector_type(8)))  int          i32x8_t;
typedef __attribute__((ext_vector_type(4)))  int          i32x4_t;

#define kB   2
#define kT   2048
#define kBT  4096      /* B*T */
#define kH   2048
#define kNH  32
#define kHD  64
#define kNKV 4
#define kKV  256       /* NKV*HD */
#define kNREP 8
#define kFF  5632

#define LDS_STRIDE 40  /* bf16 elems per LDS A-row: 32 data + 8 pad (80B) */

__device__ __forceinline__ unsigned short f2bf(float f) {
  union { float f; unsigned u; } v; v.f = f;
  unsigned r = v.u + 0x7FFFu + ((v.u >> 16) & 1u);   // round-to-nearest-even
  return (unsigned short)(r >> 16);
}
__device__ __forceinline__ float bf2f(unsigned short h) {
  union { unsigned u; float f; } v; v.u = (unsigned)h << 16;
  return v.f;
}

union V16U { uint4 u[2]; v16bf v; unsigned short s[16]; };

// ------------------------------- cast f32 -> bf16 ---------------------------
__global__ void cast_f32_bf16(const float* __restrict__ in,
                              unsigned short* __restrict__ out, int n) {
  int i = blockIdx.x * blockDim.x + threadIdx.x;
  if (i < n) out[i] = f2bf(in[i]);
}

// ------------------------------- RMSNorm ------------------------------------
__global__ void rmsnorm_f32_bf16(const float* __restrict__ x,
                                 const float* __restrict__ w,
                                 unsigned short* __restrict__ out) {
  __shared__ float red[8];
  int row = blockIdx.x;
  const float* xr = x + (size_t)row * kH;
  float ss = 0.f;
  for (int i = threadIdx.x; i < kH; i += 256) { float v = xr[i]; ss += v * v; }
  for (int off = 16; off > 0; off >>= 1) ss += __shfl_down(ss, off, 32);
  int wave = threadIdx.x >> 5, lane = threadIdx.x & 31;
  if (lane == 0) red[wave] = ss;
  __syncthreads();
  if (threadIdx.x == 0) {
    float t = 0.f;
    for (int i = 0; i < 8; i++) t += red[i];
    red[0] = rsqrtf(t / (float)kH + 1e-5f);
  }
  __syncthreads();
  float rinv = red[0];
  unsigned short* orow = out + (size_t)row * kH;
  for (int i = threadIdx.x; i < kH; i += 256) orow[i] = f2bf(xr[i] * rinv * w[i]);
}

// ------------------------------- WMMA GEMM with TDM A-staging ---------------
// C[M,N] = A[M,K](bf16,row-major) x W[N,K](bf16,row-major)^T
// block = 256 thr = 8 waves (2 M x 4 N); block tile 128x64; wave tile 64x16.
// A 128x32 tile per K-step DMA'd to LDS by the Tensor Data Mover (one wave
// issues the descriptor), double buffered; pad 16B/row -> 80B LDS row stride.
// EPI 0: store bf16.  EPI 1: store f32 with residual add.
template <int EPI>
__global__ void gemm_bf16_wmma(const unsigned short* __restrict__ A,
                               const unsigned short* __restrict__ W,
                               void* __restrict__ Out,
                               const float* __restrict__ Res,
                               int M, int N, int K) {
  __shared__ unsigned short tileA[2][128 * LDS_STRIDE];

  int wid   = threadIdx.x >> 5;
  int lane  = threadIdx.x & 31;
  int wm    = wid >> 2;                 // 0..1
  int wn    = wid & 3;                  // 0..3
  int m0    = blockIdx.x * 128 + wm * 64;
  int n0    = blockIdx.y * 64 + wn * 16;
  int lhalf = lane >> 4;                // 0|1
  int lidx  = lane & 15;

  v8f acc[4] = {};
  const int kA0 = lhalf * 8;            // A lane K-chunks: 0..7/16..23 | 8..15/24..31
  const int kA1 = 16 + lhalf * 8;
  const unsigned short* wrow = W + (size_t)(n0 + lidx) * K + lhalf * 8;

  // --- TDM descriptor issue: whole 128x32 bf16 tile -> LDS (padded rows) ---
  auto issue_tdm = [&](int buf, int k) {
    unsigned lds = (unsigned)(uintptr_t)&tileA[buf][0];
    unsigned long long ga =
        (unsigned long long)(uintptr_t)(A + (size_t)blockIdx.x * 128 * K + k);
    u32x4_t g0;
    g0[0] = 1u;                                         // count=1, user D#
    g0[1] = lds;                                        // lds_addr
    g0[2] = (unsigned)ga;                               // global_addr[31:0]
    g0[3] = (unsigned)((ga >> 32) & 0x1FFFFFFull)       // global_addr[56:32]
            | (2u << 30);                               // type=2 ("image")
    i32x8_t g1;
    g1[0] = (1 << 16)     // data_size = 2 bytes
          | (1 << 20)     // pad_enable
          | (3 << 22)     // pad_interval: 16 DWORDs (=64B row) before pad
          | (3 << 25);    // pad_amount: 4 DWORDs (=16B pad)
    g1[1] = (K & 0xFFFF) << 16;                          // tensor_dim0[15:0]
    g1[2] = ((K >> 16) & 0xFFFF) | ((M & 0xFFFF) << 16); // dim0[31:16] | dim1[15:0]
    g1[3] = ((M >> 16) & 0xFFFF) | (32 << 16);           // dim1[31:16] | tile_dim0=32
    g1[4] = 128;                                         // tile_dim1=128, tile_dim2=0
    g1[5] = K;                                           // tensor_dim0_stride[31:0]
    g1[6] = 0;                                           // stride[47:32]=0, dim1_stride lo=0
    g1[7] = 0;
    i32x4_t z4 = {0, 0, 0, 0};                           // groups 2/3 unused (2D)
    i32x8_t z8 = {0, 0, 0, 0, 0, 0, 0, 0};
    __builtin_amdgcn_tensor_load_to_lds(g0, g1, z4, z4, z8, 0);
  };

  if (wid == 0) {
    issue_tdm(0, 0);
    __builtin_amdgcn_s_wait_tensorcnt(0);
  }
  __syncthreads();

  int cur = 0;
  const unsigned short* lbaseA;
  for (int k = 0; k < K; k += 32) {
    if (wid == 0 && (k + 32) < K) issue_tdm(cur ^ 1, k + 32);

    V16U b;
    b.u[0] = *(const uint4*)(wrow + k);
    b.u[1] = *(const uint4*)(wrow + k + 16);
    __builtin_prefetch(wrow + k + 64, 0, 1);     // global_prefetch_b8

    lbaseA = &tileA[cur][(size_t)(wm * 64 + lidx) * LDS_STRIDE];
#pragma unroll
    for (int i = 0; i < 4; i++) {
      const unsigned short* lr = lbaseA + (size_t)(16 * i) * LDS_STRIDE;
      V16U a;
      a.u[0] = *(const uint4*)(lr + kA0);        // ds_load_b128
      a.u[1] = *(const uint4*)(lr + kA1);
      acc[i] = __builtin_amdgcn_wmma_f32_16x16x32_bf16(
          false, a.v, false, b.v, (short)0, acc[i], false, false);
    }

    if (wid == 0) __builtin_amdgcn_s_wait_tensorcnt(0);
    __syncthreads();
    cur ^= 1;
  }

  int col = n0 + lidx;
#pragma unroll
  for (int i = 0; i < 4; i++) {
#pragma unroll
    for (int r = 0; r < 8; r++) {
      int row = m0 + 16 * i + r + 8 * lhalf;     // C layout: lanes 16-31 = rows 8-15
      size_t idx = (size_t)row * N + col;
      float v = acc[i][r];
      if (EPI == 0) ((unsigned short*)Out)[idx] = f2bf(v);
      else          ((float*)Out)[idx] = v + Res[idx];
    }
  }
}

// ------------------------------- RoPE (in place, bf16) ----------------------
__global__ void rope_bf16(unsigned short* __restrict__ qk, int nheads) {
  int idx = blockIdx.x * blockDim.x + threadIdx.x;
  int total = kBT * nheads * (kHD / 2);
  if (idx >= total) return;
  int i   = idx & 31;                    // freq index 0..31
  int h   = (idx >> 5) % nheads;
  int row = idx / (32 * nheads);
  int t   = row % kT;
  float inv = __powf(10000.f, -(float)i / 32.f);
  float ang = (float)t * inv;
  float c = __cosf(ang), s = __sinf(ang);
  size_t base = (size_t)row * (nheads * kHD) + h * kHD + i;
  float x1 = bf2f(qk[base]);
  float x2 = bf2f(qk[base + 32]);
  qk[base]      = f2bf(x1 * c - x2 * s);
  qk[base + 32] = f2bf(x2 * c + x1 * s);
}

// --------------------------- V transpose: [BT,KV] -> [B,NKV,HD,T] -----------
__global__ void transpose_v(const unsigned short* __restrict__ v,
                            unsigned short* __restrict__ vt) {
  int idx = blockIdx.x * blockDim.x + threadIdx.x;
  int total = kB * kNKV * kHD * kT;
  if (idx >= total) return;
  int t  = idx % kT;
  int d  = (idx / kT) % kHD;
  int kh = (idx / (kT * kHD)) % kNKV;
  int b  = idx / (kT * kHD * kNKV);
  vt[idx] = v[(size_t)(b * kT + t) * kKV + kh * kHD + d];
}

// ------------------------------- Flash attention ----------------------------
// One wave per (b, head, 16-query tile). S^T = K·Q^T so softmax reductions are
// in-lane + one xor-16 shuffle; P^T C-layout feeds the P·V A-operand directly.
__global__ void attn_wmma(const unsigned short* __restrict__ Q,   // [BT,H]
                          const unsigned short* __restrict__ Kt,  // [BT,KV]
                          const unsigned short* __restrict__ Vt,  // [B,NKV,HD,T]
                          unsigned short* __restrict__ Oout) {    // [BT,H]
  int gwave = (blockIdx.x * blockDim.x + threadIdx.x) >> 5;
  int lane  = threadIdx.x & 31;
  int qtile = gwave % (kT / 16);
  int h     = (gwave / (kT / 16)) % kNH;
  int b     = gwave / ((kT / 16) * kNH);
  int kh    = h / kNREP;
  int q0    = qtile * 16;
  int lhalf = lane >> 4, lidx = lane & 15;

  const unsigned short* qrow = Q + (size_t)(b * kT + q0 + lidx) * kH + h * kHD;
  V16U qb0, qb1;
  qb0.u[0] = *(const uint4*)(qrow + lhalf * 16);
  qb0.u[1] = *(const uint4*)(qrow + lhalf * 16 + 8);
  qb1.u[0] = *(const uint4*)(qrow + 32 + lhalf * 16);
  qb1.u[1] = *(const uint4*)(qrow + 32 + lhalf * 16 + 8);

  v8f o[4] = {};
  float m = -1e30f, l = 0.f;
  const float scale = 0.125f;            // 1/sqrt(64)
  const int kA0 = lhalf * 16, kA1 = lhalf * 16 + 8;
  const unsigned short* vbase = Vt + (size_t)(b * kNKV + kh) * kHD * kT;
  int qg = q0 + lidx;

  for (int kv = 0; kv < q0 + 16; kv += 32) {
    const unsigned short* kr0 = Kt + (size_t)(b * kT + kv + lidx) * kKV + kh * kHD;
    const unsigned short* kr1 = Kt + (size_t)(b * kT + kv + 16 + lidx) * kKV + kh * kHD;
    V16U a00, a01, a10, a11;
    a00.u[0] = *(const uint4*)(kr0 + kA0);      a00.u[1] = *(const uint4*)(kr0 + kA1);
    a01.u[0] = *(const uint4*)(kr0 + 32 + kA0); a01.u[1] = *(const uint4*)(kr0 + 32 + kA1);
    a10.u[0] = *(const uint4*)(kr1 + kA0);      a10.u[1] = *(const uint4*)(kr1 + kA1);
    a11.u[0] = *(const uint4*)(kr1 + 32 + kA0); a11.u[1] = *(const uint4*)(kr1 + 32 + kA1);

    v8f st0 = {}, st1 = {};
    st0 = __builtin_amdgcn_wmma_f32_16x16x32_bf16(false, a00.v, false, qb0.v, (short)0, st0, false, false);
    st0 = __builtin_amdgcn_wmma_f32_16x16x32_bf16(false, a01.v, false, qb1.v, (short)0, st0, false, false);
    st1 = __builtin_amdgcn_wmma_f32_16x16x32_bf16(false, a10.v, false, qb0.v, (short)0, st1, false, false);
    st1 = __builtin_amdgcn_wmma_f32_16x16x32_bf16(false, a11.v, false, qb1.v, (short)0, st1, false, false);

#pragma unroll
    for (int r = 0; r < 8; r++) {
      int key0 = kv + r + 8 * lhalf;
      int key1 = kv + 16 + r + 8 * lhalf;
      float s0 = st0[r] * scale; if (key0 > qg) s0 = -1e30f;
      float s1 = st1[r] * scale; if (key1 > qg) s1 = -1e30f;
      st0[r] = s0; st1[r] = s1;
    }
    float bm = -1e30f;
#pragma unroll
    for (int r = 0; r < 8; r++) bm = fmaxf(bm, fmaxf(st0[r], st1[r]));
    bm = fmaxf(bm, __shfl_xor(bm, 16, 32));
    float mnew  = fmaxf(m, bm);
    float alpha = __expf(m - mnew);
    float ls = 0.f;
#pragma unroll
    for (int r = 0; r < 8; r++) {
      st0[r] = __expf(st0[r] - mnew);
      st1[r] = __expf(st1[r] - mnew);
      ls += st0[r] + st1[r];
    }
    ls += __shfl_xor(ls, 16, 32);
    l = l * alpha + ls;
    m = mnew;

#pragma unroll
    for (int r = 0; r < 8; r++) {
      float ar = __shfl(alpha, r + 8 * lhalf, 32);
#pragma unroll
      for (int j = 0; j < 4; j++) o[j][r] *= ar;
    }

    V16U pa;
#pragma unroll
    for (int r = 0; r < 8; r++) { pa.s[r] = f2bf(st0[r]); pa.s[8 + r] = f2bf(st1[r]); }

#pragma unroll
    for (int j = 0; j < 4; j++) {
      const unsigned short* vr = vbase + (size_t)(j * 16 + lidx) * kT + kv + lhalf * 8;
      V16U vb;
      vb.u[0] = *(const uint4*)(vr);
      vb.u[1] = *(const uint4*)(vr + 16);
      o[j] = __builtin_amdgcn_wmma_f32_16x16x32_bf16(
          false, pa.v, false, vb.v, (short)0, o[j], false, false);
    }
  }

  float linv = 1.f / l;
#pragma unroll
  for (int r = 0; r < 8; r++) {
    float lr = __shfl(linv, r + 8 * lhalf, 32);
    int row = b * kT + q0 + r + 8 * lhalf;
#pragma unroll
    for (int j = 0; j < 4; j++) {
      int col = h * kHD + j * 16 + lidx;
      Oout[(size_t)row * kH + col] = f2bf(o[j][r] * lr);
    }
  }
}

// ------------------------------- SiLU(gate)*up ------------------------------
__global__ void silu_mul_bf16(const unsigned short* __restrict__ g,
                              const unsigned short* __restrict__ u,
                              unsigned short* __restrict__ out, int n) {
  int i = blockIdx.x * blockDim.x + threadIdx.x;
  if (i >= n) return;
  float x = bf2f(g[i]);
  float s = x / (1.f + __expf(-x));
  out[i] = f2bf(s * bf2f(u[i]));
}

// ---------------------------------------------------------------------------
extern "C" void kernel_launch(void* const* d_in, const int* in_sizes, int n_in,
                              void* d_out, int out_size, void* d_ws, size_t ws_size,
                              hipStream_t stream) {
  (void)in_sizes; (void)n_in; (void)out_size; (void)ws_size;
  const float* x     = (const float*)d_in[0];
  const float* wq    = (const float*)d_in[1];
  const float* wk    = (const float*)d_in[2];
  const float* wv    = (const float*)d_in[3];
  const float* wo    = (const float*)d_in[4];
  const float* wgate = (const float*)d_in[5];
  const float* wup   = (const float*)d_in[6];
  const float* wdown = (const float*)d_in[7];
  const float* ln1   = (const float*)d_in[8];
  const float* ln2   = (const float*)d_in[9];
  float* out = (float*)d_out;

  char* ws = (char*)d_ws;
  size_t off = 0;
  auto alloc = [&](size_t bytes) {
    char* p = ws + off;
    off = (off + bytes + 255) & ~(size_t)255;
    return p;
  };
  unsigned short* wq_bf  = (unsigned short*)alloc((size_t)kH * kH * 2);
  unsigned short* wk_bf  = (unsigned short*)alloc((size_t)kKV * kH * 2);
  unsigned short* wv_bf  = (unsigned short*)alloc((size_t)kKV * kH * 2);
  unsigned short* wo_bf  = (unsigned short*)alloc((size_t)kH * kH * 2);
  unsigned short* wg_bf  = (unsigned short*)alloc((size_t)kFF * kH * 2);
  unsigned short* wu_bf  = (unsigned short*)alloc((size_t)kFF * kH * 2);
  unsigned short* wd_bf  = (unsigned short*)alloc((size_t)kH * kFF * 2);
  unsigned short* hn_bf  = (unsigned short*)alloc((size_t)kBT * kH * 2);
  unsigned short* q_bf   = (unsigned short*)alloc((size_t)kBT * kH * 2);
  unsigned short* k_bf   = (unsigned short*)alloc((size_t)kBT * kKV * 2);
  unsigned short* v_bf   = (unsigned short*)alloc((size_t)kBT * kKV * 2);
  unsigned short* vt_bf  = (unsigned short*)alloc((size_t)kBT * kKV * 2);
  unsigned short* ao_bf  = (unsigned short*)alloc((size_t)kBT * kH * 2);
  float*          x1_f   = (float*)alloc((size_t)kBT * kH * 4);
  unsigned short* hn2_bf = (unsigned short*)alloc((size_t)kBT * kH * 2);
  unsigned short* gt_bf  = (unsigned short*)alloc((size_t)kBT * kFF * 2);
  unsigned short* up_bf  = (unsigned short*)alloc((size_t)kBT * kFF * 2);
  unsigned short* gu_bf  = (unsigned short*)alloc((size_t)kBT * kFF * 2);

  auto cast = [&](const float* src, unsigned short* dst, int n) {
    cast_f32_bf16<<<(n + 255) / 256, 256, 0, stream>>>(src, dst, n);
  };
  cast(wq, wq_bf, kH * kH);
  cast(wk, wk_bf, kKV * kH);
  cast(wv, wv_bf, kKV * kH);
  cast(wo, wo_bf, kH * kH);
  cast(wgate, wg_bf, kFF * kH);
  cast(wup, wu_bf, kFF * kH);
  cast(wdown, wd_bf, kH * kFF);

  // ---- attention block ----
  rmsnorm_f32_bf16<<<kBT, 256, 0, stream>>>(x, ln1, hn_bf);

  dim3 blk(256);
  gemm_bf16_wmma<0><<<dim3(kBT / 128, kH / 64), blk, 0, stream>>>(
      hn_bf, wq_bf, q_bf, nullptr, kBT, kH, kH);
  gemm_bf16_wmma<0><<<dim3(kBT / 128, kKV / 64), blk, 0, stream>>>(
      hn_bf, wk_bf, k_bf, nullptr, kBT, kKV, kH);
  gemm_bf16_wmma<0><<<dim3(kBT / 128, kKV / 64), blk, 0, stream>>>(
      hn_bf, wv_bf, v_bf, nullptr, kBT, kKV, kH);

  rope_bf16<<<(kBT * kNH * 32 + 255) / 256, 256, 0, stream>>>(q_bf, kNH);
  rope_bf16<<<(kBT * kNKV * 32 + 255) / 256, 256, 0, stream>>>(k_bf, kNKV);
  transpose_v<<<(kBT * kKV + 255) / 256, 256, 0, stream>>>(v_bf, vt_bf);

  attn_wmma<<<(kB * kNH * (kT / 16) * 32) / 256, 256, 0, stream>>>(
      q_bf, k_bf, vt_bf, ao_bf);

  gemm_bf16_wmma<1><<<dim3(kBT / 128, kH / 64), blk, 0, stream>>>(
      ao_bf, wo_bf, x1_f, x, kBT, kH, kH);

  // ---- MLP block ----
  rmsnorm_f32_bf16<<<kBT, 256, 0, stream>>>(x1_f, ln2, hn2_bf);

  gemm_bf16_wmma<0><<<dim3(kBT / 128, kFF / 64), blk, 0, stream>>>(
      hn2_bf, wg_bf, gt_bf, nullptr, kBT, kFF, kH);
  gemm_bf16_wmma<0><<<dim3(kBT / 128, kFF / 64), blk, 0, stream>>>(
      hn2_bf, wu_bf, up_bf, nullptr, kBT, kFF, kH);

  silu_mul_bf16<<<(kBT * kFF + 255) / 256, 256, 0, stream>>>(
      gt_bf, up_bf, gu_bf, kBT * kFF);

  gemm_bf16_wmma<1><<<dim3(kBT / 128, kH / 64), blk, 0, stream>>>(
      gu_bf, wd_bf, out, x1_f, kBT, kH, kFF);
}
